// CensusLoss_38895223833176
// MI455X (gfx1250) — compile-verified, matrix operand order checked by reading, not run on hardware
//
#include <hip/hip_runtime.h>
#include <cstdint>

typedef __attribute__((ext_vector_type(16))) _Float16 v16h;
typedef __attribute__((ext_vector_type(8)))  float    v8f;

#define IMG_H 512
#define IMG_W 512
#define HWSZ  (IMG_H * IMG_W)
#define TILE  32
#define HALO  3
#define REG   (TILE + 2 * HALO)   // 38
#define LDSS  (REG + 2)           // 40, pad to avoid bank conflicts

__device__ __forceinline__ int reflect_idx(int i) {
    // jnp.pad mode='reflect' (no edge repeat): i<0 -> -i ; i>=N -> 2N-2-i
    i = (i < 0) ? -i : i;
    i = (i >= IMG_H) ? (2 * IMG_H - 2 - i) : i;
    return i;
}

__global__ __launch_bounds__(256) void census_kernel(const float* __restrict__ pred,
                                                     const float* __restrict__ targ,
                                                     unsigned int* __restrict__ acc) {
    __shared__ float GP[REG * LDSS];
    __shared__ float GT[REG * LDSS];
    __shared__ float partial[16];

    const int t   = threadIdx.x;
    const int b   = blockIdx.z;
    const int g0r = blockIdx.y * TILE - HALO;
    const int g0c = blockIdx.x * TILE - HALO;
    const size_t base = (size_t)b * 3 * HWSZ;

    // Stage gray tiles (pred & target) into LDS; coalesced global reads,
    // each RGB pixel of this tile region is read exactly once.
    for (int e = t; e < REG * REG; e += 256) {
        int lr = e / REG;
        int lc = e - lr * REG;
        int gr = reflect_idx(g0r + lr);
        int gc = reflect_idx(g0c + lc);
        size_t idx = base + (size_t)gr * IMG_W + gc;
        float rp = pred[idx], gp = pred[idx + HWSZ], bp = pred[idx + 2 * HWSZ];
        float rt = targ[idx], gt = targ[idx + HWSZ], bt = targ[idx + 2 * HWSZ];
        GP[lr * LDSS + lc] = 0.299f * rp + 0.587f * gp + 0.114f * bp;
        GT[lr * LDSS + lc] = 0.299f * rt + 0.587f * gt + 0.114f * bt;
    }
    __syncthreads();

    // Each thread handles 4 output pixels: (ty0 + 8r, tx), r = 0..3.
    const int tx  = t & 31;
    const int ty0 = t >> 5;
    int count = 0;
    #pragma unroll
    for (int r = 0; r < 4; ++r) {
        int ly = ty0 + r * 8;
        const float* rowP = &GP[ly * LDSS + tx];
        const float* rowT = &GT[ly * LDSS + tx];
        float cp = rowP[HALO * LDSS + HALO];
        float ct = rowT[HALO * LDSS + HALO];
        #pragma unroll
        for (int i = 0; i < 7; ++i) {
            #pragma unroll
            for (int j = 0; j < 7; ++j) {
                if (i == 3 && j == 3) continue;
                float np = rowP[i * LDSS + j];
                float nt = rowT[i * LDSS + j];
                count += ((cp > np) != (ct > nt)) ? 1 : 0;
            }
        }
    }

    // ---- Per-wave reduction via one v_wmma_f32_16x16x32_f16 ----
    // A: every lane broadcasts its count into all 16 of its f16 elements.
    // B: all ones. D[m,n] = sum_k A[m,k]; column 0 of D lives in lanes 0 and 16,
    // and sum_m D[m,0] = 16 * sum_over_lanes(count). Exact: count <= 192.
    _Float16 ch = (_Float16)((float)count);
    v16h a, bo;
    #pragma unroll
    for (int k = 0; k < 16; ++k) { a[k] = ch; bo[k] = (_Float16)1.0f; }
    v8f cz = {};
    v8f d = __builtin_amdgcn_wmma_f32_16x16x32_f16(
        /*neg_a=*/false, a, /*neg_b=*/false, bo,
        /*c_mod=*/(short)0, cz, /*reuse_a=*/false, /*reuse_b=*/false);
    float s = d[0] + d[1] + d[2] + d[3] + d[4] + d[5] + d[6] + d[7];

    const int lane = t & 31;
    const int wave = t >> 5;
    if (lane == 0)  partial[wave * 2 + 0] = s;
    if (lane == 16) partial[wave * 2 + 1] = s;
    __syncthreads();

    if (t == 0) {
        float tot = 0.0f;
        #pragma unroll
        for (int k = 0; k < 16; ++k) tot += partial[k];
        // tot = 16 * (workgroup mismatch count); exact integer-valued f32.
        unsigned int u = (unsigned int)(tot * (1.0f / 16.0f) + 0.5f);
        atomicAdd(acc, u);
    }
}

__global__ void zero_acc_kernel(unsigned int* acc) {
    if (threadIdx.x == 0 && blockIdx.x == 0) *acc = 0u;
}

__global__ void finalize_kernel(const unsigned int* __restrict__ acc,
                                float* __restrict__ out) {
    if (threadIdx.x == 0 && blockIdx.x == 0) {
        // mean over 8 * 48 * 512 * 512 = 100663296 elements
        out[0] = (float)(*acc) * (1.0f / 100663296.0f);
    }
}

extern "C" void kernel_launch(void* const* d_in, const int* in_sizes, int n_in,
                              void* d_out, int out_size, void* d_ws, size_t ws_size,
                              hipStream_t stream) {
    const float* pred = (const float*)d_in[0];
    const float* targ = (const float*)d_in[1];
    unsigned int* acc = (unsigned int*)d_ws;

    const int B = in_sizes[0] / (3 * HWSZ);   // 8

    zero_acc_kernel<<<1, 32, 0, stream>>>(acc);

    dim3 grid(IMG_W / TILE, IMG_H / TILE, B); // 16 x 16 x 8 = 2048 WGs
    census_kernel<<<grid, 256, 0, stream>>>(pred, targ, acc);

    finalize_kernel<<<1, 32, 0, stream>>>(acc, (float*)d_out);
}
